// SparseEnhancer_26508538151606
// MI455X (gfx1250) — compile-verified, matrix operand order checked by reading, not run on hardware
//
#include <hip/hip_runtime.h>
#include <hip/hip_bf16.h>

#define B_      8192
#define D_      128
#define NITEM   16384
#define HIDDEN  32
#define TOPK    10
#define TEMPINV 10.0f        // 1 / 0.1
#define NEG_INF -9.0e9f
#define EPSV    1e-12f

#define ROWS    32           // rows per workgroup strip
#define KP      128          // K panel length (f16 elems)
#define KPAN    (NITEM/KP)   // 128 panels
#define NCH     (B_/128)     // 64 column chunks of 128 cols

typedef _Float16 v8h  __attribute__((ext_vector_type(8)));
typedef _Float16 v16h __attribute__((ext_vector_type(16)));
typedef float    v8f  __attribute__((ext_vector_type(8)));
typedef int      v4i  __attribute__((ext_vector_type(4)));

// ---- CDNA5 async global->LDS copy (probe for builtin, fall back to VGPR path)
#if defined(__has_builtin)
# if __has_builtin(__builtin_amdgcn_global_load_async_to_lds_b128)
#  define HAVE_ASYNC_LDS 1
# endif
#endif
#ifndef HAVE_ASYNC_LDS
# define HAVE_ASYNC_LDS 0
#endif

typedef __attribute__((address_space(1))) v4i gv4i;
typedef __attribute__((address_space(3))) v4i lv4i;

__device__ __forceinline__ void copy16_g2l(const void* g, void* l) {
#if HAVE_ASYNC_LDS
    __builtin_amdgcn_global_load_async_to_lds_b128(
        (gv4i*)(void*)g, (lv4i*)(void*)l, 0, 0);
#else
    *(uint4*)l = *(const uint4*)g;
#endif
}

__device__ __forceinline__ void wait_async_copies() {
#if HAVE_ASYNC_LDS
# if defined(__has_builtin) && __has_builtin(__builtin_amdgcn_s_wait_asynccnt)
    __builtin_amdgcn_s_wait_asynccnt(0);
# else
    asm volatile("s_wait_asynccnt 0" ::: "memory");
# endif
#endif
}

// ---------------------------------------------------------------- init / max
__global__ void k_init(int* maxlen) {
    if (threadIdx.x == 0) *maxlen = 0;
}

__global__ void k_maxlen(const int* __restrict__ real_len, int* maxlen) {
    int i = blockIdx.x * blockDim.x + threadIdx.x;
    if (i < B_) atomicMax(maxlen, real_len[i]);
}

// ---------------------------------------------------------------- alpha MLP
__global__ void k_alpha(const int* __restrict__ real_len, const int* __restrict__ maxlen,
                        const float* __restrict__ W1, const float* __restrict__ b1,
                        const float* __restrict__ W2, const float* __restrict__ b2,
                        float* __restrict__ alpha) {
    int b = blockIdx.x * blockDim.x + threadIdx.x;
    if (b >= B_) return;
    float ml = fmaxf((float)(*maxlen), 1.0f);
    float s  = 1.0f - (float)real_len[b] / ml;
    float acc = b2[0];
#pragma unroll
    for (int j = 0; j < HIDDEN; j++) {
        float h = fmaxf(s * W1[j] + b1[j], 0.0f);
        acc += W2[j] * h;
    }
    alpha[b] = 1.0f / (1.0f + __expf(-acc));
}

// ---------------------------------------------------------------- row norm -> f16
__global__ void k_rownorm(const float* __restrict__ t, _Float16* __restrict__ tn) {
    int b = blockIdx.x;
    __shared__ float red[256];
    const float* row = t + (size_t)b * NITEM;
    float s = 0.0f;
    for (int i = threadIdx.x; i < NITEM; i += 256) { float v = row[i]; s += v * v; }
    red[threadIdx.x] = s;
    __syncthreads();
    for (int off = 128; off > 0; off >>= 1) {
        if (threadIdx.x < off) red[threadIdx.x] += red[threadIdx.x + off];
        __syncthreads();
    }
    float inv = 1.0f / fmaxf(sqrtf(red[0]), EPSV);
    _Float16* orow = tn + (size_t)b * NITEM;
    for (int i = threadIdx.x; i < NITEM; i += 256)
        orow[i] = (_Float16)(row[i] * inv);
}

// ---------------------------------------------------------------- fused GEMM + topk + blend
__global__ __launch_bounds__(256, 1) void k_strip(
    const _Float16* __restrict__ tn, const float* __restrict__ z,
    const float* __restrict__ alpha, float* __restrict__ out,
    float* __restrict__ partial) {

    __shared__ __align__(16) _Float16 Ap[ROWS * KP];   //  8 KB
    __shared__ __align__(16) _Float16 Bp[128 * KP];    // 32 KB (reused as merge buf)
    __shared__ float simbuf[ROWS * 128];               // 16 KB
    __shared__ float red[256];
    __shared__ float wrow[ROWS * TOPK];
    __shared__ int   irow[ROWS * TOPK];

    const int tid  = threadIdx.x;
    const int lane = tid & 31;
    const int w    = tid >> 5;            // wave id 0..7 -> 16-col subtile
    const int r0   = blockIdx.x * ROWS;   // row strip base

    // running per-thread top-k: row = tid&31, column segment = tid>>5
    float tv[TOPK]; int ti[TOPK];
#pragma unroll
    for (int k = 0; k < TOPK; k++) { tv[k] = -3.0e38f; ti[k] = 0; }
    const int prow = tid & 31;
    const int pseg = tid >> 5;

    const int m    = lane & 15;          // WMMA lane row/col
    const int half = lane >> 4;          // WMMA lane half (K grouping)

    for (int c = 0; c < NCH; c++) {
        const int cbase = c * 128;
        v8f acc0 = {}, acc1 = {};
        for (int kp = 0; kp < KPAN; kp++) {
            const int kbase = kp * KP;
            __syncthreads();
            // A panel: 32 rows x 128 f16 = 512 x 16B chunks (2 per thread)
#pragma unroll
            for (int i = 0; i < 2; i++) {
                int idx = tid + i * 256;
                int arow = idx >> 4, ac = idx & 15;
                copy16_g2l(tn + (size_t)(r0 + arow) * NITEM + kbase + ac * 8,
                           Ap + arow * KP + ac * 8);
            }
            // B panel: 128 rows x 128 f16 = 2048 x 16B chunks (8 per thread)
#pragma unroll
            for (int i = 0; i < 8; i++) {
                int idx = tid + i * 256;
                int brow = idx >> 4, bc = idx & 15;
                copy16_g2l(tn + (size_t)(cbase + brow) * NITEM + kbase + bc * 8,
                           Bp + brow * KP + bc * 8);
            }
            wait_async_copies();
            __syncthreads();
#pragma unroll
            for (int kk = 0; kk < KP / 32; kk++) {
                const _Float16* bbase = Bp + (w * 16 + m) * KP + kk * 32 + half * 8;
                v8h blo = *(const v8h*)bbase;
                v8h bhi = *(const v8h*)(bbase + 16);
                v16h b = __builtin_shufflevector(blo, bhi, 0,1,2,3,4,5,6,7,8,9,10,11,12,13,14,15);

                const _Float16* abase0 = Ap + m * KP + kk * 32 + half * 8;
                v8h a0lo = *(const v8h*)abase0;
                v8h a0hi = *(const v8h*)(abase0 + 16);
                v16h a0 = __builtin_shufflevector(a0lo, a0hi, 0,1,2,3,4,5,6,7,8,9,10,11,12,13,14,15);

                const _Float16* abase1 = Ap + (16 + m) * KP + kk * 32 + half * 8;
                v8h a1lo = *(const v8h*)abase1;
                v8h a1hi = *(const v8h*)(abase1 + 16);
                v16h a1 = __builtin_shufflevector(a1lo, a1hi, 0,1,2,3,4,5,6,7,8,9,10,11,12,13,14,15);

                acc0 = __builtin_amdgcn_wmma_f32_16x16x32_f16(
                        false, a0, false, b, (short)0, acc0, false, false);
                acc1 = __builtin_amdgcn_wmma_f32_16x16x32_f16(
                        false, a1, false, b, (short)0, acc1, false, false);
            }
        }
        // dump 32x16 tile pair -> simbuf with diagonal mask
        {
            int n  = lane & 15;
            int rb = (lane >> 4) * 8;
            int gc = cbase + w * 16 + n;
#pragma unroll
            for (int i = 0; i < 8; i++) {
                int gr0 = r0 + rb + i;
                float v0 = acc0[i];
                if (gr0 == gc) v0 = NEG_INF;
                simbuf[(rb + i) * 128 + w * 16 + n] = v0;
                int gr1 = r0 + 16 + rb + i;
                float v1 = acc1[i];
                if (gr1 == gc) v1 = NEG_INF;
                simbuf[(16 + rb + i) * 128 + w * 16 + n] = v1;
            }
        }
        __syncthreads();
        // online partial top-k over this chunk (16 cols per thread)
#pragma unroll
        for (int j = 0; j < 16; j++) {
            int col = pseg * 16 + j;
            float v = simbuf[prow * 128 + col];
            int gid = cbase + col;
            if (v > tv[TOPK - 1]) {
                tv[TOPK - 1] = v; ti[TOPK - 1] = gid;
#pragma unroll
                for (int k = TOPK - 1; k > 0; k--) {
                    if (tv[k] > tv[k - 1]) {
                        float fv = tv[k]; tv[k] = tv[k - 1]; tv[k - 1] = fv;
                        int   fi = ti[k]; ti[k] = ti[k - 1]; ti[k - 1] = fi;
                    }
                }
            }
        }
    }

    // merge 8 partial top-10s per row (overlay merge buffers on Bp: 20 KB < 32 KB)
    float* mergeV = (float*)Bp;                        // 32*8*10 floats = 10 KB
    int*   mergeI = ((int*)Bp) + ROWS * 8 * TOPK;      // next 10 KB
#pragma unroll
    for (int k = 0; k < TOPK; k++) {
        mergeV[(prow * 8 + pseg) * TOPK + k] = tv[k];
        mergeI[(prow * 8 + pseg) * TOPK + k] = ti[k];
    }
    __syncthreads();
    if (tid < ROWS) {
        float fv[TOPK]; int fi[TOPK];
#pragma unroll
        for (int k = 0; k < TOPK; k++) { fv[k] = -3.0e38f; fi[k] = 0; }
        for (int s = 0; s < 8; s++) {
            for (int k = 0; k < TOPK; k++) {
                float v  = mergeV[(tid * 8 + s) * TOPK + k];
                int   id = mergeI[(tid * 8 + s) * TOPK + k];
                if (v > fv[TOPK - 1]) {
                    fv[TOPK - 1] = v; fi[TOPK - 1] = id;
                    for (int q = TOPK - 1; q > 0; q--) {
                        if (fv[q] > fv[q - 1]) {
                            float t1 = fv[q]; fv[q] = fv[q - 1]; fv[q - 1] = t1;
                            int   t2 = fi[q]; fi[q] = fi[q - 1]; fi[q - 1] = t2;
                        }
                    }
                }
            }
        }
        float mx = fv[0], sum = 0.0f, e[TOPK];
        for (int k = 0; k < TOPK; k++) { e[k] = __expf((fv[k] - mx) * TEMPINV); sum += e[k]; }
        float inv = 1.0f / sum;
        for (int k = 0; k < TOPK; k++) { wrow[tid * TOPK + k] = e[k] * inv; irow[tid * TOPK + k] = fi[k]; }
    }
    __syncthreads();

    // neighbor aggregation, blend, output, loss partial (16 dims per thread)
    {
        int orow = tid >> 3;          // 0..31
        int dseg = tid & 7;           // 0..7 -> 16 dims
        int gr   = r0 + orow;
        const float* zr = z + (size_t)gr * D_ + dseg * 16;
        float zi[16], nz[16];
#pragma unroll
        for (int d = 0; d < 16; d++) { zi[d] = zr[d]; nz[d] = 0.0f; }
#pragma unroll
        for (int k = 0; k < TOPK; k++) {
            float wv = wrow[orow * TOPK + k];
            const float* zn = z + (size_t)irow[orow * TOPK + k] * D_ + dseg * 16;
#pragma unroll
            for (int d = 0; d < 16; d++) nz[d] += wv * zn[d];
        }
        float a  = alpha[gr];
        float* o = out + (size_t)gr * D_ + dseg * 16;
        float sq = 0.0f;
#pragma unroll
        for (int d = 0; d < 16; d++) {
            float zt = (1.0f - a) * zi[d] + a * nz[d];
            o[d] = zt;
            float df = zt - zi[d];
            sq += df * df;
        }
        red[tid] = sq;
    }
    __syncthreads();
    for (int off = 128; off > 0; off >>= 1) {
        if (tid < off) red[tid] += red[tid + off];
        __syncthreads();
    }
    if (tid == 0) partial[blockIdx.x] = red[0];
}

// ---------------------------------------------------------------- final loss reduce
__global__ void k_final(const float* __restrict__ partial, float* __restrict__ out_scalar) {
    __shared__ float red[256];
    red[threadIdx.x] = partial[threadIdx.x];   // B_/ROWS == 256 partials
    __syncthreads();
    for (int off = 128; off > 0; off >>= 1) {
        if (threadIdx.x < off) red[threadIdx.x] += red[threadIdx.x + off];
        __syncthreads();
    }
    if (threadIdx.x == 0) *out_scalar = red[0] / (float)((size_t)B_ * D_);
}

// ---------------------------------------------------------------- launch
extern "C" void kernel_launch(void* const* d_in, const int* in_sizes, int n_in,
                              void* d_out, int out_size, void* d_ws, size_t ws_size,
                              hipStream_t stream) {
    (void)in_sizes; (void)n_in; (void)out_size; (void)ws_size;
    const float* z        = (const float*)d_in[0];
    const float* t        = (const float*)d_in[1];
    const int*   real_len = (const int*)  d_in[2];
    const float* W1       = (const float*)d_in[3];
    const float* b1       = (const float*)d_in[4];
    const float* W2       = (const float*)d_in[5];
    const float* b2       = (const float*)d_in[6];
    float* out = (float*)d_out;

    char* ws          = (char*)ws_size ? (char*)d_ws : (char*)d_ws;
    ws                = (char*)d_ws;
    int*   maxlen     = (int*)ws;                       // 4 B
    float* partial    = (float*)(ws + 64);              // 256 floats
    float* alpha      = (float*)(ws + 4096);            // 8192 floats
    _Float16* tn      = (_Float16*)(ws + (1 << 20));    // 8192*16384 f16 = 256 MB

    k_init  <<<1, 1, 0, stream>>>(maxlen);
    k_maxlen<<<(B_ + 255) / 256, 256, 0, stream>>>(real_len, maxlen);
    k_alpha <<<(B_ + 255) / 256, 256, 0, stream>>>(real_len, maxlen, W1, b1, W2, b2, alpha);
    k_rownorm<<<B_, 256, 0, stream>>>(t, tn);
    k_strip <<<B_ / ROWS, 256, 0, stream>>>(tn, z, alpha, out, partial);
    k_final <<<1, 256, 0, stream>>>(partial, out + (size_t)B_ * D_);
}
